// HungarianMatcher_49684181680873
// MI455X (gfx1250) — matrix operand order repaired and checked by reading, not compiled, not run on hardware
//
#include <hip/hip_runtime.h>
#include <hip/hip_bf16.h>
#include <math.h>

typedef __attribute__((ext_vector_type(16))) _Float16 v16h;
typedef __attribute__((ext_vector_type(8)))  _Float16 v8h;
typedef __attribute__((ext_vector_type(8)))  float    v8f;

#define W_CLASS 2.0f
#define W_BBOX  5.0f
#define W_GIOU  2.0f

// ---------------------------------------------------------------------------
// Kernel 0: prep. A16 = sigmoid(logits) as f16, zero-padded to (Ppad x Kpad).
//           B16 = one-hot(labels) as f16, K-contiguous per gt column,
//                 zero-padded to (Mpad x Kpad).
// Branch-light, fully vector-parallel; removes the scalarized exp/div mess
// from the WMMA kernel's fragment build.
// ---------------------------------------------------------------------------
__global__ void hm_prep(const float* __restrict__ logits,
                        const int*   __restrict__ labels,
                        _Float16* __restrict__ A16,
                        _Float16* __restrict__ B16,
                        int P, int C, int M, int Kpad, int nA, int nB)
{
  int t = blockIdx.x * blockDim.x + threadIdx.x;
  if (t < nA) {
    int p = t / Kpad, k = t - p * Kpad;
    float v = 0.0f;
    if (p < P && k < C)
      v = 1.0f / (1.0f + __expf(-logits[(size_t)p * C + k]));
    A16[t] = (_Float16)v;
  } else if (t < nA + nB) {
    int t2 = t - nA;
    int n = t2 / Kpad, k = t2 - n * Kpad;
    B16[t2] = (_Float16)((n < M && labels[n] == k) ? 1.0f : 0.0f);
  }
}

// ---------------------------------------------------------------------------
// Kernel 1: fused cost matrix. One wave32 per 16x16 output tile.
//   cost_class via v_wmma_f32_16x16x32_f16:  probs (P x Kpad) @ onehot (Kpad x M)
//   A-frag (16-bit A 16x32, ISA 7.12.2): per lane, halves 0-7 = K[ha..ha+8),
//   halves 8-15 = K[ha+16..ha+24) with ha = kb + (lane<16 ? 0 : 8)
//   -> two contiguous 16B loads. B-frag: same K order as A:
//   halves 0-7 = K[hb..hb+8), 8-15 = K[hb+16..hb+24), hb = ha -> two 16B loads.
//   bbox L1 + IoU fused in the epilogue via the C/D f32 lane layout.
// Stores C^T (M x P) since the LSAP runs on the transpose (P >= M path).
// ---------------------------------------------------------------------------
__global__ __launch_bounds__(32)
void hm_cost_wmma(const _Float16* __restrict__ A16,   // (Ppad, Kpad)
                  const _Float16* __restrict__ B16,   // (Mpad, Kpad) col-major-in-K
                  const float4*   __restrict__ pb4,   // (P,) xyxy
                  const float4*   __restrict__ gb4,   // (M,) xyxy
                  float* __restrict__ Ct,             // (M, P)
                  int P, int M, int Kpad)
{
  const int lane = threadIdx.x;                  // 0..31
  const int tp   = blockIdx.x;                   // pred tile (WMMA M dim)
  const int tm   = blockIdx.y;                   // gt tile   (WMMA N dim)
  const _Float16* aRow = A16 + (size_t)(tp * 16 + (lane & 15)) * Kpad;
  const int n_g = tm * 16 + (lane & 15);
  const _Float16* bCol = B16 + (size_t)n_g * Kpad;

  v8f acc = {};
#pragma unroll 3
  for (int kb = 0; kb < Kpad; kb += 32) {
    const int ha = kb + ((lane < 16) ? 0 : 8);
    v8h alo = *(const v8h*)(aRow + ha);
    v8h ahi = *(const v8h*)(aRow + ha + 16);
    v16h a = __builtin_shufflevector(alo, ahi, 0, 1, 2, 3, 4, 5, 6, 7,
                                     8, 9, 10, 11, 12, 13, 14, 15);
    const int hb = kb + ((lane < 16) ? 0 : 8);
    v8h blo = *(const v8h*)(bCol + hb);
    v8h bhi = *(const v8h*)(bCol + hb + 16);
    v16h b = __builtin_shufflevector(blo, bhi, 0, 1, 2, 3, 4, 5, 6, 7,
                                     8, 9, 10, 11, 12, 13, 14, 15);
    acc = __builtin_amdgcn_wmma_f32_16x16x32_f16(false, a, false, b,
                                                 (short)0, acc, false, false);
  }

  if (n_g >= M) return;   // no WMMA below this point; divergence is safe

  const float4 gb = gb4[n_g];
  const float area_b = (gb.z - gb.x) * (gb.w - gb.y);
#pragma unroll
  for (int r = 0; r < 8; ++r) {
    // C/D layout: lane L, VGPR r -> M = r + 8*(L>=16), N = L%16
    int p = tp * 16 + r + ((lane < 16) ? 0 : 8);
    if (p >= P) continue;
    float prob = acc[r];                 // = sigmoid(logits[p, labels[n_g]])
    const float4 pb = pb4[p];
    float l1 = fabsf(pb.x - gb.x) + fabsf(pb.y - gb.y)
             + fabsf(pb.z - gb.z) + fabsf(pb.w - gb.w);
    float area_a = (pb.z - pb.x) * (pb.w - pb.y);
    float iw = fmaxf(fminf(pb.z, gb.z) - fmaxf(pb.x, gb.x), 0.0f);
    float ih = fmaxf(fminf(pb.w, gb.w) - fmaxf(pb.y, gb.y), 0.0f);
    float inter = iw * ih;
    float iou = inter / (area_a + area_b - inter);
    Ct[(size_t)n_g * P + p] = W_CLASS * (-prob) + W_BBOX * l1 + W_GIOU * (-iou);
  }
}

// ---------------------------------------------------------------------------
// Kernel 2: Jonker-Volgenant LSAP on C^T (NR=64 rows, NC=900 cols).
// Single workgroup, 1024 threads (32 wave32s). Whole cost matrix + all solver
// state lives in LDS (~249 KB) -- fits only in CDNA5's 320 KB WGP LDS.
// Cost matrix staged via per-lane global_load_async_to_lds_b128 (ASYNCcnt).
// Argmin tie-breaks on lowest column index to match np.argmin exactly.
// ---------------------------------------------------------------------------
__global__ __launch_bounds__(1024)
void hm_hungarian(const float* __restrict__ Ct, int* __restrict__ out,
                  int NR, int NC)
{
  extern __shared__ float smem[];
  float* costS  = smem;                       // NR*NC
  float* vS     = costS + (size_t)NR * NC;    // NC
  float* shortS = vS + NC;                    // NC
  float* uS     = shortS + NC;                // NR
  int* pathS    = (int*)(uS + NR);            // NC
  int* row4colS = pathS + NC;                 // NC
  int* SCm      = row4colS + NC;              // NC
  int* SRm      = SCm + NC;                   // NR
  int* col4rowS = SRm + NR;                   // NR
  float* redV   = (float*)(col4rowS + NR);    // 32
  int* redI     = (int*)(redV + 32);          // 32
  int* s_bi     = redI + 32;                  // [0]=current row i, [1]=sink
  float* s_bf   = (float*)(s_bi + 2);         // [0]=min_val

  const int tid = threadIdx.x;
  const int nth = blockDim.x;
  const int lane = tid & 31;
  const int wid = tid >> 5;
  const int nwaves = nth >> 5;

#if defined(__gfx1250__)
  {
    // Async DMA the cost matrix into LDS: 16B per lane per issue, tracked by
    // ASYNCcnt; each thread drains its own count, then the barrier in the row
    // loop publishes LDS to the whole workgroup.
    const size_t nbytes = (size_t)NR * NC * sizeof(float);
    const char* gsrc = (const char*)Ct;
    unsigned lds_base =
        (unsigned)(size_t)(__attribute__((address_space(3))) char*)(void*)costS;
    for (size_t off = (size_t)tid * 16; off + 16 <= nbytes;
         off += (size_t)nth * 16) {
      unsigned ldsa = lds_base + (unsigned)off;
      unsigned long long ga = (unsigned long long)(gsrc + off);
      asm volatile("global_load_async_to_lds_b128 %0, %1, off"
                   :: "v"(ldsa), "v"(ga) : "memory");
    }
    for (size_t off = (nbytes & ~(size_t)15) / 4 + tid; off < nbytes / 4;
         off += nth)
      costS[off] = Ct[off];
    asm volatile("s_wait_asynccnt 0x0" ::: "memory");
  }
#else
  for (int idx = tid; idx < NR * NC; idx += nth) costS[idx] = Ct[idx];
#endif
  for (int j = tid; j < NC; j += nth) { vS[j] = 0.0f; row4colS[j] = -1; }
  for (int t = tid; t < NR; t += nth) { uS[t] = 0.0f; col4rowS[t] = -1; }

  for (int cur_row = 0; cur_row < NR; ++cur_row) {
    for (int j = tid; j < NC; j += nth) { shortS[j] = INFINITY; pathS[j] = -1; SCm[j] = 0; }
    for (int t = tid; t < NR; t += nth) SRm[t] = 0;
    if (tid == 0) { s_bi[0] = cur_row; s_bi[1] = -1; s_bf[0] = 0.0f; }

    float minv = 0.0f; int sink = -1;
    for (;;) {
      __syncthreads();                        // publish s_bi/s_bf + SCm/shortS
      const int i = s_bi[0];
      sink = s_bi[1];
      minv = s_bf[0];
      if (sink >= 0) break;
      if (tid == 0) SRm[i] = 1;

      float bv = INFINITY; int bj = 0x7FFFFFFF;
      if (tid < NC && SCm[tid] == 0) {
        float r = minv + costS[(size_t)i * NC + tid] - uS[i] - vS[tid];
        if (r < shortS[tid]) { shortS[tid] = r; pathS[tid] = i; }
        bv = shortS[tid]; bj = tid;
      }
      // wave32 argmin (first-index tie-break), then cross-wave reduce in LDS
#pragma unroll
      for (int off = 16; off > 0; off >>= 1) {
        float ov = __shfl_down(bv, off, 32);
        int   oj = __shfl_down(bj, off, 32);
        if (ov < bv || (ov == bv && oj < bj)) { bv = ov; bj = oj; }
      }
      if (lane == 0) { redV[wid] = bv; redI[wid] = bj; }
      __syncthreads();
      if (tid == 0) {
        float b0 = redV[0]; int j0 = redI[0];
        for (int w = 1; w < nwaves; ++w)
          if (redV[w] < b0 || (redV[w] == b0 && redI[w] < j0)) { b0 = redV[w]; j0 = redI[w]; }
        SCm[j0] = 1;
        s_bf[0] = b0;
        if (row4colS[j0] < 0) s_bi[1] = j0;   // sink found
        else                  s_bi[0] = row4colS[j0];
      }
    }

    // dual updates (read col4row BEFORE augmentation, as in reference)
    for (int t = tid; t < NR; t += nth) {
      if (t == cur_row)  uS[t] += minv;
      else if (SRm[t])   uS[t] += minv - shortS[col4rowS[t]];
    }
    for (int j = tid; j < NC; j += nth)
      if (SCm[j]) vS[j] += shortS[j] - minv;
    __syncthreads();
    if (tid == 0) {                           // augment alternating path (<= NR hops)
      int j = sink;
      for (;;) {
        int ii = pathS[j];
        row4colS[j] = ii;
        int tmp = col4rowS[ii];
        col4rowS[ii] = j;
        j = tmp;
        if (ii == cur_row) break;
      }
    }
    __syncthreads();
  }

  // P >= M path of the reference: pred = col4row, gt = arange, sorted by pred.
  // Matched pred indices are unique -> rank by counting smaller entries.
  if (tid < NR) {
    int myp = col4rowS[tid];
    int rank = 0;
    for (int m2 = 0; m2 < NR; ++m2) rank += (col4rowS[m2] < myp) ? 1 : 0;
    out[rank]      = myp;   // pred_idx (int32)
    out[NR + rank] = tid;   // gt_idx   (int32)
  }
}

// ---------------------------------------------------------------------------
extern "C" void kernel_launch(void* const* d_in, const int* in_sizes, int n_in,
                              void* d_out, int out_size, void* d_ws, size_t ws_size,
                              hipStream_t stream) {
  (void)n_in; (void)out_size; (void)ws_size;
  const float* logits = (const float*)d_in[0];   // (P, C) f32
  const float* pboxes = (const float*)d_in[1];   // (P, 4) f32
  const float* gboxes = (const float*)d_in[2];   // (M, 4) f32
  const int*   labels = (const int*)d_in[3];     // (M,)   i32

  const int P = in_sizes[1] / 4;                 // 900
  const int M = in_sizes[3];                     // 64
  const int C = in_sizes[0] / P;                 // 91
  const int Kpad = ((C + 31) / 32) * 32;         // 96
  const int Ppad = ((P + 15) / 16) * 16;         // 912
  const int Mpad = ((M + 15) / 16) * 16;         // 64

  // Workspace layout: Ct (M*P f32) | A16 (Ppad*Kpad f16) | B16 (Mpad*Kpad f16)
  char* ws = (char*)d_ws;
  float*    Ct  = (float*)ws;
  size_t ctB = (size_t)M * P * sizeof(float);
  ctB = (ctB + 15) & ~(size_t)15;
  _Float16* A16 = (_Float16*)(ws + ctB);
  size_t aB = (size_t)Ppad * Kpad * sizeof(_Float16);
  aB = (aB + 15) & ~(size_t)15;
  _Float16* B16 = (_Float16*)(ws + ctB + aB);

  const int nA = Ppad * Kpad, nB = Mpad * Kpad;
  const int tot = nA + nB;
  hm_prep<<<(tot + 255) / 256, 256, 0, stream>>>(logits, labels, A16, B16,
                                                 P, C, M, Kpad, nA, nB);

  dim3 g1(Ppad / 16, Mpad / 16);
  hm_cost_wmma<<<g1, 32, 0, stream>>>(A16, B16, (const float4*)pboxes,
                                      (const float4*)gboxes, Ct, P, M, Kpad);

  // LDS: cost(M*P) + v,shortest(P) + u(M) floats; path,row4col,SC(P) + SR,col4row(M)
  //      ints + 64 reduce slots + 3 broadcast words -> ~249.4 KB < 320 KB WGP LDS
  size_t smem = sizeof(float) * ((size_t)M * P + 2 * (size_t)P + M)
              + sizeof(int)   * (3 * (size_t)P + 2 * (size_t)M)
              + (32 + 32 + 3) * sizeof(int) + 16;
  hm_hungarian<<<1, 1024, smem, stream>>>(Ct, (int*)d_out, M, P);
}
